// Attention_87771951661871
// MI455X (gfx1250) — matrix operand order repaired and checked by reading, not compiled, not run on hardware
//
#include <hip/hip_runtime.h>
#include <hip/hip_bf16.h>

// ---------------------------------------------------------------------------
// Problem constants (from the reference)
// ---------------------------------------------------------------------------
#define BB   16
#define HH   56
#define WW   56
#define CC   128
#define NN   (HH * WW)          // 3136
#define NHEAD 8
#define HD   16                 // head dim == WMMA tile dim
#define LKV  240                // kv sequence length (token1+token2+token3)
#define MTOT (BB * NN)          // 50176 rows for big GEMMs
#define MSEQ (BB * LKV)         // 3840 rows for kv GEMMs

typedef __attribute__((ext_vector_type(16))) __bf16 v16bf;
typedef __attribute__((ext_vector_type(8)))  __bf16 v8bf;
typedef __attribute__((ext_vector_type(8)))  float  v8f;

// ---------------------------------------------------------------------------
// fp32 -> bf16 conversion
// ---------------------------------------------------------------------------
__global__ void f2bf_kernel(const float* __restrict__ in, __bf16* __restrict__ out, long n) {
  long i = (long)blockIdx.x * blockDim.x + threadIdx.x;
  if (i < n) out[i] = (__bf16)in[i];
}

__global__ void addcvt_kernel(const float* __restrict__ a, const float* __restrict__ b,
                              __bf16* __restrict__ out, long n) {
  long i = (long)blockIdx.x * blockDim.x + threadIdx.x;
  if (i < n) out[i] = (__bf16)(a[i] + b[i]);
}

// ---------------------------------------------------------------------------
// Stable ascending argsort of mask per batch via rank counting (matches
// jnp.argsort stability). One block per batch, mask cached in LDS.
// ---------------------------------------------------------------------------
__global__ __launch_bounds__(256)
void sort_kernel(const float* __restrict__ mask, int* __restrict__ idx) {
  __shared__ float sm[NN];
  const int b = blockIdx.x;
  for (int i = threadIdx.x; i < NN; i += blockDim.x) sm[i] = mask[(long)b * NN + i];
  __syncthreads();
  for (int i = threadIdx.x; i < NN; i += blockDim.x) {
    const float mi = sm[i];
    int rank = 0;
    for (int j = 0; j < NN; ++j) {
      const float mj = sm[j];
      rank += (mj < mi) || (mj == mi && j < i);
    }
    idx[(long)b * NN + rank] = i;
  }
}

// ---------------------------------------------------------------------------
// Build seq1 (learned f-reductions over sorted token groups) and seq2
// (top-mask tokens), both emitted in bf16 (B*240, 128).
// grid = (240, B), block = 128 (one channel per thread)
// ---------------------------------------------------------------------------
__global__ __launch_bounds__(128)
void seq_kernel(const float* __restrict__ x, const int* __restrict__ idx,
                const float* __restrict__ f1w, const float* __restrict__ f1b,
                const float* __restrict__ f2w, const float* __restrict__ f2b,
                const float* __restrict__ f3w, const float* __restrict__ f3b,
                __bf16* __restrict__ seq1, __bf16* __restrict__ seq2) {
  const int t = blockIdx.x, b = blockIdx.y, c = threadIdx.x;
  int f, base; const float* fw; float fb;
  if (t < 16)       { f = 49; base = t * 49;              fw = f1w; fb = f1b[0]; }
  else if (t < 128) { f = 14; base = 784  + (t - 16) * 14; fw = f2w; fb = f2b[0]; }
  else              { f = 7;  base = 2352 + (t - 128) * 7; fw = f3w; fb = f3b[0]; }
  float acc = fb;
  for (int j = 0; j < f; ++j) {
    const int tok = idx[(long)b * NN + base + j];
    acc += x[((long)b * NN + tok) * CC + c] * fw[j];
  }
  seq1[((long)b * LKV + t) * CC + c] = (__bf16)acc;
  const int tok2 = idx[(long)b * NN + (NN - LKV) + t];
  seq2[((long)b * LKV + t) * CC + c] = (__bf16)x[((long)b * NN + tok2) * CC + c];
}

// ---------------------------------------------------------------------------
// Generic WMMA GEMM: out[M,Ncols] = A[M,K](bf16) * W[Ncols,K]^T(bf16) + bias.
// One 16x16 output tile per wave; 8 waves per block cover 128 output cols.
// A lane layout (16-bit A 16x32):  lanes 0-15: K = {0..7,16..23}, M = lane;
//                                  lanes16-31: K = {8..15,24..31}, M = lane-16.
// B lane layout (16-bit B 32x16):  lanes 0-15: N = lane,   K = {0..7,16..23};
//                                  lanes16-31: N = lane-16, K = {8..15,24..31}.
// D layout: VGPR r -> row = (lane<16 ? r : r+8), col = lane&15.
// ---------------------------------------------------------------------------
template <typename OutT>
__global__ __launch_bounds__(256)
void gemm_bf16_kernel(const __bf16* __restrict__ A, const __bf16* __restrict__ W,
                      const float* __restrict__ bias, OutT* __restrict__ out,
                      int M, int K, int Ncols) {
  const int lane  = threadIdx.x & 31;
  const int wave  = threadIdx.x >> 5;
  const int half  = lane >> 4;
  const int l16   = lane & 15;
  const int mtile = blockIdx.x;
  const int ntile = blockIdx.y * 8 + wave;
  if (ntile * 16 >= Ncols) return;

  const __bf16* arow = A + (long)(mtile * 16 + l16) * K;
  const __bf16* wrow = W + (long)(ntile * 16 + l16) * K;

  v8f acc = {};
  for (int kc = 0; kc < K; kc += 32) {
    if (kc + 32 < K) {  // gfx1250 global_prefetch_b8 for the next K chunk
      __builtin_prefetch(arow + kc + 32 + half * 8, 0, 3);
      __builtin_prefetch(wrow + kc + 32 + half * 16, 0, 3);
    }
    const v8bf a0 = *(const v8bf*)(arow + kc + half * 8);
    const v8bf a1 = *(const v8bf*)(arow + kc + 16 + half * 8);
    const v8bf b0 = *(const v8bf*)(wrow + kc + half * 8);
    const v8bf b1 = *(const v8bf*)(wrow + kc + 16 + half * 8);
    v16bf a, bm;
#pragma unroll
    for (int e = 0; e < 8; ++e) { a[e] = a0[e]; a[e + 8] = a1[e]; bm[e] = b0[e]; bm[e + 8] = b1[e]; }
    acc = __builtin_amdgcn_wmma_f32_16x16x32_bf16(false, a, false, bm, (short)0, acc, false, false);
  }

  const int  ncol = ntile * 16 + l16;
  const float bv  = bias ? bias[ncol] : 0.0f;
#pragma unroll
  for (int r = 0; r < 8; ++r) {
    const int row = mtile * 16 + half * 8 + r;
    out[(long)row * Ncols + ncol] = (OutT)(acc[r] + bv);
  }
}

// ---------------------------------------------------------------------------
// Rearrange kv1/kv2 GEMM outputs:
//   Kd : head-major rows (B*240, 128)  -> contiguous loads for q.k^T B-tiles
//   Vt : transposed (B, NH, 16, 240)   -> contiguous loads for p.v B-tiles
// k heads 0-3 <- kv1[:, 0:64], heads 4-7 <- kv2[:, 0:64]; v from cols 64:128.
// ---------------------------------------------------------------------------
__global__ __launch_bounds__(128)
void kv_rearrange_kernel(const __bf16* __restrict__ kv1, const __bf16* __restrict__ kv2,
                         __bf16* __restrict__ Kd, __bf16* __restrict__ Vt) {
  const long r = blockIdx.x;            // r = b*240 + t
  const int  b = (int)(r / LKV), t = (int)(r % LKV);
  const int  c = threadIdx.x;
  const int  h = c >> 4, d = c & 15;
  const __bf16* src = (h < 4) ? kv1 : kv2;
  const int hh = (h < 4) ? h : h - 4;
  Kd[r * CC + c] = src[r * CC + hh * 16 + d];
  Vt[(((long)b * NHEAD + h) * HD + d) * LKV + t] = src[r * CC + 64 + hh * 16 + d];
}

// ---------------------------------------------------------------------------
// lepe depthwise 3x3 conv (SAME / zero pad) over the linear output held as
// (B, N, C). grid = (3136, B), block = 128.
// ---------------------------------------------------------------------------
__global__ __launch_bounds__(128)
void lepe_conv_kernel(const float* __restrict__ lin, const float* __restrict__ cw,
                      const float* __restrict__ cb, float* __restrict__ out) {
  const int n = blockIdx.x, b = blockIdx.y, c = threadIdx.x;
  const int y = n / WW, x = n % WW;
  float acc = cb[c];
#pragma unroll
  for (int dy = -1; dy <= 1; ++dy)
#pragma unroll
    for (int dx = -1; dx <= 1; ++dx) {
      const int yy = y + dy, xx = x + dx;
      if (yy >= 0 && yy < HH && xx >= 0 && xx < WW)
        acc += lin[((long)b * NN + yy * WW + xx) * CC + c] * cw[c * 9 + (dy + 1) * 3 + (dx + 1)];
    }
  out[((long)b * NN + n) * CC + c] = acc;
}

// ---------------------------------------------------------------------------
// Attention: per (b, h, 16-row tile) one wave.
// Phase 1: S = q_tile (16x16, K padded to 32) x k^T  -> 15 WMMAs -> 16x240.
// Softmax across 240 per row (cross-lane max/sum within 16-lane halves).
// Phase 2: P (staged to LDS, zero-padded to 16x256) x V -> 8 WMMAs -> 16x16.
// V is consumed from the transposed Vt (B,NH,16,240): each lane's B slice is
// two 8-bf16 runs (K=kc+8h.., kc+16+8h..); the K=240..255 pad is exactly (kc==224, upper half).
// ---------------------------------------------------------------------------
__global__ __launch_bounds__(32)
void attn_kernel(const __bf16* __restrict__ Q, const __bf16* __restrict__ Km,
                 const __bf16* __restrict__ Vt, float* __restrict__ Out) {
  __shared__ __bf16 Plds[16 * 256];
  const int lane = threadIdx.x & 31;
  const int half = lane >> 4, l16 = lane & 15;
  const int mtile = blockIdx.x, h = blockIdx.y, b = blockIdx.z;

  // A = q tile, K dim 16 zero-padded to 32 (elements 8..15 carry K>=16 -> 0)
  v16bf a = {};
  {
    const __bf16* qrow = Q + ((long)(b * NN + mtile * 16 + l16)) * CC + h * 16 + half * 8;
    const v8bf q0 = *(const v8bf*)qrow;
#pragma unroll
    for (int e = 0; e < 8; ++e) a[e] = q0[e];
  }

  v8f s[15];
#pragma unroll
  for (int j = 0; j < 15; ++j) {
    v16bf bm = {};
    {  // elements 0..7 hold K=8*half+e; elements 8..15 (K>=16) are zero pad
      const __bf16* krow = Km + ((long)(b * LKV + j * 16 + l16)) * CC + h * 16 + half * 8;
      const v8bf k0 = *(const v8bf*)krow;
#pragma unroll
      for (int e = 0; e < 8; ++e) { bm[e] = k0[e]; }
    }
    v8f z = {};
    s[j] = __builtin_amdgcn_wmma_f32_16x16x32_bf16(false, a, false, bm, (short)0, z, false, false);
  }

  // softmax over 240 columns for each of the 8 rows this half-wave owns
  const float scale = 0.25f;  // hd^-0.5 = 16^-0.5
#pragma unroll
  for (int r = 0; r < 8; ++r) {
    float m = -1e30f;
#pragma unroll
    for (int j = 0; j < 15; ++j) m = fmaxf(m, s[j][r]);
#pragma unroll
    for (int off = 1; off < 16; off <<= 1) m = fmaxf(m, __shfl_xor(m, off, 16));
    float sum = 0.0f;
#pragma unroll
    for (int j = 0; j < 15; ++j) { const float e = __expf((s[j][r] - m) * scale); s[j][r] = e; sum += e; }
#pragma unroll
    for (int off = 1; off < 16; off <<= 1) sum += __shfl_xor(sum, off, 16);
    const float inv = 1.0f / sum;
#pragma unroll
    for (int j = 0; j < 15; ++j) s[j][r] *= inv;
  }

  // stage P to LDS in row-major (16 x 256), zero the pad columns 240..255
  for (int i = lane; i < 256; i += 32) Plds[(i >> 4) * 256 + 240 + (i & 15)] = (__bf16)0.0f;
#pragma unroll
  for (int j = 0; j < 15; ++j)
#pragma unroll
    for (int r = 0; r < 8; ++r)
      Plds[(half * 8 + r) * 256 + j * 16 + l16] = (__bf16)s[j][r];
  __syncthreads();

  // Phase 2: out = P (16x256) x V (256x16, rows >=240 zero)
  v8f o = {};
  for (int kc = 0; kc < 256; kc += 32) {
    const __bf16* prow = &Plds[l16 * 256 + kc + half * 8];
    const v8bf p0 = *(const v8bf*)prow;
    const v8bf p1 = *(const v8bf*)(prow + 16);
    v8bf v0 = {}, v1 = {};
    {  // only v1 of the final chunk (K=240..255) is pad
      const __bf16* vrow = Vt + (((long)(b * NHEAD + h)) * HD + l16) * LKV + kc + half * 8;
      v0 = *(const v8bf*)vrow;
      if (kc != 224) v1 = *(const v8bf*)(vrow + 16);
    }
    v16bf a2, b2;
#pragma unroll
    for (int e = 0; e < 8; ++e) { a2[e] = p0[e]; a2[e + 8] = p1[e]; b2[e] = v0[e]; b2[e + 8] = v1[e]; }
    o = __builtin_amdgcn_wmma_f32_16x16x32_bf16(false, a2, false, b2, (short)0, o, false, false);
  }

  const long orow = (long)(b * NN + mtile * 16 + half * 8);
#pragma unroll
  for (int r = 0; r < 8; ++r)
    Out[(orow + r) * CC + h * 16 + l16] = o[r];
}

// ---------------------------------------------------------------------------
// Host-side launcher
// ---------------------------------------------------------------------------
extern "C" void kernel_launch(void* const* d_in, const int* in_sizes, int n_in,
                              void* d_out, int out_size, void* d_ws, size_t ws_size,
                              hipStream_t stream) {
  (void)in_sizes; (void)n_in; (void)out_size; (void)ws_size;

  const float* x        = (const float*)d_in[0];
  const float* mask     = (const float*)d_in[1];
  // d_in[2] = H, d_in[3] = W (compile-time constants here)
  const float* W_q      = (const float*)d_in[4];
  const float* W_kv1    = (const float*)d_in[5];
  const float* W_kv2    = (const float*)d_in[6];
  const float* f1_w     = (const float*)d_in[7];
  const float* f1_b     = (const float*)d_in[8];
  const float* f2_w     = (const float*)d_in[9];
  const float* f2_b     = (const float*)d_in[10];
  const float* f3_w     = (const float*)d_in[11];
  const float* f3_b     = (const float*)d_in[12];
  const float* lepe_w   = (const float*)d_in[13];
  const float* lepe_b   = (const float*)d_in[14];
  const float* conv_w   = (const float*)d_in[15];
  const float* conv_b   = (const float*)d_in[16];
  const float* proj_w   = (const float*)d_in[17];
  const float* proj_b   = (const float*)d_in[18];
  float* out = (float*)d_out;

  // workspace carve-up (256B aligned)
  char* ws = (char*)d_ws;
  size_t off = 0;
  auto take = [&](size_t bytes) { size_t r = off; off += (bytes + 255) & ~(size_t)255; return r; };
  const size_t o_xbf  = take((size_t)MTOT * CC * 2);   // x in bf16
  const size_t o_qbf  = take((size_t)MTOT * CC * 2);   // q in bf16 (later reused for sum_bf16)
  const size_t o_wq   = take((size_t)CC * CC * 2);
  const size_t o_wk1  = take((size_t)CC * CC * 2);
  const size_t o_wk2  = take((size_t)CC * CC * 2);
  const size_t o_wlin = take((size_t)CC * CC * 2);
  const size_t o_wprj = take((size_t)CC * CC * 2);
  const size_t o_idx  = take((size_t)BB * NN * 4);
  const size_t o_s1   = take((size_t)MSEQ * CC * 2);
  const size_t o_s2   = take((size_t)MSEQ * CC * 2);
  const size_t o_kv1  = take((size_t)MSEQ * CC * 2);
  const size_t o_kv2  = take((size_t)MSEQ * CC * 2);
  const size_t o_k    = take((size_t)MSEQ * CC * 2);
  const size_t o_vt   = take((size_t)BB * NHEAD * HD * LKV * 2);
  const size_t o_bigA = take((size_t)MTOT * CC * 4);   // lin_f32, then attn_f32
  const size_t o_bigB = take((size_t)MTOT * CC * 4);   // lepe_f32

  __bf16* xbf  = (__bf16*)(ws + o_xbf);
  __bf16* qbf  = (__bf16*)(ws + o_qbf);
  __bf16* wqb  = (__bf16*)(ws + o_wq);
  __bf16* wk1b = (__bf16*)(ws + o_wk1);
  __bf16* wk2b = (__bf16*)(ws + o_wk2);
  __bf16* wlnb = (__bf16*)(ws + o_wlin);
  __bf16* wpjb = (__bf16*)(ws + o_wprj);
  int*    idx  = (int*)(ws + o_idx);
  __bf16* s1b  = (__bf16*)(ws + o_s1);
  __bf16* s2b  = (__bf16*)(ws + o_s2);
  __bf16* kv1b = (__bf16*)(ws + o_kv1);
  __bf16* kv2b = (__bf16*)(ws + o_kv2);
  __bf16* kb   = (__bf16*)(ws + o_k);
  __bf16* vtb  = (__bf16*)(ws + o_vt);
  float*  bigA = (float*)(ws + o_bigA);
  float*  bigB = (float*)(ws + o_bigB);
  __bf16* sumb = qbf;  // reuse q region once q is consumed

  const long nx = (long)MTOT * CC;
  const long nw = (long)CC * CC;

  // 1) fp32 -> bf16 conversions
  f2bf_kernel<<<dim3((nx + 255) / 256), 256, 0, stream>>>(x, xbf, nx);
  f2bf_kernel<<<dim3((nw + 255) / 256), 256, 0, stream>>>(W_q,    wqb,  nw);
  f2bf_kernel<<<dim3((nw + 255) / 256), 256, 0, stream>>>(W_kv1,  wk1b, nw);
  f2bf_kernel<<<dim3((nw + 255) / 256), 256, 0, stream>>>(W_kv2,  wk2b, nw);
  f2bf_kernel<<<dim3((nw + 255) / 256), 256, 0, stream>>>(lepe_w, wlnb, nw);
  f2bf_kernel<<<dim3((nw + 255) / 256), 256, 0, stream>>>(proj_w, wpjb, nw);

  // 2) stable argsort of mask per batch
  sort_kernel<<<dim3(BB), 256, 0, stream>>>(mask, idx);

  // 3) seq1 / seq2 construction
  seq_kernel<<<dim3(LKV, BB), 128, 0, stream>>>(x, idx, f1_w, f1_b, f2_w, f2_b,
                                                f3_w, f3_b, s1b, s2b);

  // 4) big GEMMs: lepe-linear (f32 out) and q (bf16 out)
  gemm_bf16_kernel<float><<<dim3(MTOT / 16, 1), 256, 0, stream>>>(
      xbf, wlnb, lepe_b, bigA, MTOT, CC, CC);
  gemm_bf16_kernel<__bf16><<<dim3(MTOT / 16, 1), 256, 0, stream>>>(
      xbf, wqb, nullptr, qbf, MTOT, CC, CC);

  // 5) kv GEMMs + head rearrange (K head-major rows, V transposed per head)
  gemm_bf16_kernel<__bf16><<<dim3(MSEQ / 16, 1), 256, 0, stream>>>(
      s1b, wk1b, nullptr, kv1b, MSEQ, CC, CC);
  gemm_bf16_kernel<__bf16><<<dim3(MSEQ / 16, 1), 256, 0, stream>>>(
      s2b, wk2b, nullptr, kv2b, MSEQ, CC, CC);
  kv_rearrange_kernel<<<dim3(MSEQ), 128, 0, stream>>>(kv1b, kv2b, kb, vtb);

  // 6) lepe depthwise conv: bigA (lin) -> bigB (lepe)
  lepe_conv_kernel<<<dim3(NN, BB), 128, 0, stream>>>(bigA, conv_w, conv_b, bigB);

  // 7) attention -> bigA (overwrites lin, which is no longer needed)
  attn_kernel<<<dim3(NN / 16, NHEAD, BB), 32, 0, stream>>>(qbf, kb, vtb, bigA);

  // 8) (attn + lepe) -> bf16, then final projection into d_out
  addcvt_kernel<<<dim3((nx + 255) / 256), 256, 0, stream>>>(bigA, bigB, sumb, nx);
  gemm_bf16_kernel<float><<<dim3(MTOT / 16, 1), 256, 0, stream>>>(
      sumb, wpjb, proj_b, out, MTOT, CC, CC);
}
